// CentroidAwareVoxelization_8839042695571
// MI455X (gfx1250) — matrix operand order repaired and checked by reading, not compiled, not run on hardware
//
#include <hip/hip_runtime.h>

// ---------------- problem constants (fixed by setup_inputs) ----------------
#define Bb   8
#define Nn   16384
#define Dd   256
#define Vv   4096
#define BN_  (Bb * Nn)   // 131072 points
#define Ss   (Bb * Vv)   // 32768 segments

// ---------------- d_out offsets (floats), concatenated in return order -----
#define OUT_FEAT 0ull                 // (8,16384,256) = 33554432
#define OUT_NORM 33554432ull          // (8,16384,3)   = 393216
#define OUT_CENT 33947648ull          // (32768,3)     = 98304
#define OUT_CNT  34045952ull          // (32768,)      = 32768
#define OUT_EMB  34078720ull          // (131072,256)  = 33554432
#define OUT_BID  67633152ull          // (8,16384)     = 131072

// ---------------- workspace offsets (floats) -------------------------------
// h1 region (131072x256) is consumed by GEMM2 before the aliases are used.
#define WS_H1    0ull                 // 33554432
#define WS_AGGP  0ull                 //  9437184 (32768x288)  [alias in h1]
#define WS_WF1P  9437184ull           //    73728 (288x256)    [alias in h1]
#define WS_Y3    9510912ull           //  8388608 (32768x256)  [alias in h1]
#define WS_H2    17899520ull          //  8388608 (32768x256)  [alias in h1]
#define WS_AGG   33554432ull          //  8519680 (32768x260)
#define WS_Y4    42074112ull          //  8388608 (32768x256)
#define WS_ST    50462720ull          //     2048 (4 stages x {sum256,sq256})

typedef __attribute__((ext_vector_type(16))) __bf16 v16bf;
typedef __attribute__((ext_vector_type(8)))  float  v8f;

// ------------- bf16 helpers (manual RNE, union bit casts) ------------------
static __device__ __forceinline__ __bf16 f2bf(float f) {
  union { float f; unsigned u; } a; a.f = f;
  unsigned r = a.u + 0x7FFFu + ((a.u >> 16) & 1u);
  union { unsigned short s; __bf16 b; } o; o.s = (unsigned short)(r >> 16);
  return o.b;
}
static __device__ __forceinline__ float bf2f(__bf16 b) {
  union { unsigned short s; __bf16 b; } i; i.b = b;
  union { unsigned u; float f; } o; o.u = ((unsigned)i.s) << 16;
  return o.f;
}
static __device__ __forceinline__ float gelu_exact(float x) {
  return 0.5f * x * (1.0f + erff(x * 0.70710678118654752f));
}

// ---------------- K1: per-point scatter of count + xyz sum -----------------
__global__ void k_scatter1(const float* __restrict__ pts,
                           const int* __restrict__ vox,
                           float* __restrict__ cnt, float* __restrict__ sums) {
  int p = blockIdx.x * 256 + threadIdx.x;
  if (p >= BN_) return;
  int b = p >> 14;                       // N = 16384
  int seg = b * Vv + vox[p];
  atomicAdd(&cnt[seg], 1.0f);
  atomicAdd(&sums[seg * 3 + 0], pts[p * 3 + 0]);
  atomicAdd(&sums[seg * 3 + 1], pts[p * 3 + 1]);
  atomicAdd(&sums[seg * 3 + 2], pts[p * 3 + 2]);
}

// ---------------- K2: centroids = sums / max(cnt,1) (in place) -------------
__global__ void k_centfin(float* __restrict__ cent, const float* __restrict__ cnt) {
  int s = blockIdx.x * 256 + threadIdx.x;
  if (s >= Ss) return;
  float d = fmaxf(cnt[s], 1.0f);
  cent[s * 3 + 0] /= d; cent[s * 3 + 1] /= d; cent[s * 3 + 2] /= d;
}

// ------- K3: norm = p - centroid; padded_norm out; y1 = norm @ W1 ----------
// one wave per row; each lane produces 8 of the 256 columns
__global__ void k_norm_l1(const float* __restrict__ pts,
                          const int* __restrict__ vox,
                          const float* __restrict__ cent,
                          const float* __restrict__ W1,
                          float* __restrict__ pnorm_out,
                          float* __restrict__ y1) {
  int row  = blockIdx.x * 8 + (threadIdx.x >> 5);
  int lane = threadIdx.x & 31;
  if (row >= BN_) return;
  int b = row >> 14, n = row & (Nn - 1);
  int seg = b * Vv + vox[row];
  float nx = pts[row * 3 + 0] - cent[seg * 3 + 0];
  float ny = pts[row * 3 + 1] - cent[seg * 3 + 1];
  float nz = pts[row * 3 + 2] - cent[seg * 3 + 2];
  if (lane < 3) {
    float c = (lane == 0) ? nx : (lane == 1) ? ny : nz;
    pnorm_out[(size_t)row * 3 + lane] = (n < Vv) ? c : 0.0f;
  }
  size_t rb = (size_t)row * Dd;
  #pragma unroll
  for (int j = 0; j < 8; ++j) {
    int d = lane * 8 + j;
    y1[rb + d] = nx * W1[d] + ny * W1[Dd + d] + nz * W1[2 * Dd + d];
  }
}

// ---------------- column stats: sum & sumsq over M rows --------------------
__global__ void k_colstats(const float* __restrict__ Y, int M,
                           float* __restrict__ sum, float* __restrict__ sq) {
  int c = threadIdx.x;                               // 256 columns
  int rpc = (M + gridDim.x - 1) / gridDim.x;
  int r0 = blockIdx.x * rpc;
  int r1 = min(M, r0 + rpc);
  float s = 0.0f, q = 0.0f;
  for (int r = r0; r < r1; ++r) {
    float v = Y[(size_t)r * Dd + c];
    s += v; q += v * v;
  }
  atomicAdd(&sum[c], s); atomicAdd(&sq[c], q);
}

// ---------------- BN (training stats) + exact GELU, float4 -----------------
__global__ void k_bngelu(const float* __restrict__ Y, float* __restrict__ H,
                         const float* __restrict__ g, const float* __restrict__ bi,
                         const float* __restrict__ sum, const float* __restrict__ sq,
                         int M) {
  size_t i = (size_t)blockIdx.x * 256 + threadIdx.x;   // float4 index
  size_t n4 = (size_t)M * (Dd / 4);
  if (i >= n4) return;
  float4 y = ((const float4*)Y)[i];
  int c0 = (int)((i * 4) & (Dd - 1));
  float invM = 1.0f / (float)M;
  float o[4] = { y.x, y.y, y.z, y.w };
  #pragma unroll
  for (int j = 0; j < 4; ++j) {
    int c = c0 + j;
    float m = sum[c] * invM;
    float v = sq[c] * invM - m * m;
    float inv = rsqrtf(v + 1e-5f);
    o[j] = gelu_exact((o[j] - m) * inv * g[c] + bi[c]);
  }
  ((float4*)H)[i] = make_float4(o[0], o[1], o[2], o[3]);
}

// ---- WMMA GEMM: C[MxN] = A[MxK] @ B[KxN], N=256, split-bf16 (3 products) ---
// block = 128 threads = 4 waves; wave owns 16 rows x all 256 cols.
__global__ __launch_bounds__(128) void k_gemm(const float* __restrict__ A,
                                              const float* __restrict__ Bm,
                                              float* __restrict__ C,
                                              int K) {
  __shared__ float Bs[32 * 257];
  int tid = threadIdx.x, wave = tid >> 5, lane = tid & 31;
  int mLocal = lane & 15, half = lane >> 4;
  int rowBase = blockIdx.x * 64 + wave * 16;
  const float* Arow = A + (size_t)(rowBase + mLocal) * K;

  v8f acc[16];
  #pragma unroll
  for (int t = 0; t < 16; ++t)
    #pragma unroll
    for (int j = 0; j < 8; ++j) acc[t][j] = 0.0f;

  for (int kk = 0; kk < K; kk += 32) {
    // ---- A fragment: 16x32 bf16, ISA layout; two contiguous 8-runs / lane
    float av[16];
    int k0 = kk + half * 8;
    *(float4*)&av[0]  = *(const float4*)&Arow[k0];
    *(float4*)&av[4]  = *(const float4*)&Arow[k0 + 4];
    *(float4*)&av[8]  = *(const float4*)&Arow[k0 + 16];
    *(float4*)&av[12] = *(const float4*)&Arow[k0 + 20];
    v16bf ahi, alo;
    #pragma unroll
    for (int i = 0; i < 16; ++i) {
      __bf16 h = f2bf(av[i]);
      ahi[i] = h;
      alo[i] = f2bf(av[i] - bf2f(h));
    }
    // ---- stage B k-slab (32x256) into LDS, padded stride 257
    __syncthreads();
    for (int e = tid; e < 32 * 64; e += 128) {
      int kr = e >> 6, nc = (e & 63) * 4;
      float4 v = *(const float4*)&Bm[(size_t)(kk + kr) * Dd + nc];
      Bs[kr * 257 + nc + 0] = v.x; Bs[kr * 257 + nc + 1] = v.y;
      Bs[kr * 257 + nc + 2] = v.z; Bs[kr * 257 + nc + 3] = v.w;
    }
    __syncthreads();
    // ---- 16 n-tiles, 3 WMMAs each (hi*hi + hi*lo + lo*hi)
    #pragma unroll 4
    for (int nt = 0; nt < 16; ++nt) {
      int n = nt * 16 + mLocal;
      v16bf bhi, blo;
      #pragma unroll
      for (int i = 0; i < 16; ++i) {
        float bv = Bs[(half * 8 + i + (i >= 8 ? 8 : 0)) * 257 + n];
        __bf16 h = f2bf(bv);
        bhi[i] = h;
        blo[i] = f2bf(bv - bf2f(h));
      }
      acc[nt] = __builtin_amdgcn_wmma_f32_16x16x32_bf16(false, ahi, false, bhi,
                                                        (short)0, acc[nt], false, false);
      acc[nt] = __builtin_amdgcn_wmma_f32_16x16x32_bf16(false, ahi, false, blo,
                                                        (short)0, acc[nt], false, false);
      acc[nt] = __builtin_amdgcn_wmma_f32_16x16x32_bf16(false, alo, false, bhi,
                                                        (short)0, acc[nt], false, false);
    }
  }
  // ---- store: lane holds col (lane%16); rows half*8 + j in vgpr j
  #pragma unroll
  for (int nt = 0; nt < 16; ++nt)
    #pragma unroll
    for (int j = 0; j < 8; ++j)
      C[(size_t)(rowBase + half * 8 + j) * Dd + nt * 16 + mLocal] = acc[nt][j];
}

// ---------- scatter [xyz | emb] into per-segment accumulator ---------------
// one wave per row: lanes 0-2 add xyz, each lane adds 8 embedding columns
__global__ void k_scatter2(const float* __restrict__ pts,
                           const int* __restrict__ vox,
                           const float* __restrict__ emb,
                           float* __restrict__ agg) {
  int row  = blockIdx.x * 8 + (threadIdx.x >> 5);
  int lane = threadIdx.x & 31;
  if (row >= BN_) return;
  int b = row >> 14;
  int seg = b * Vv + vox[row];
  float* dst = agg + (size_t)seg * 260;
  if (lane < 3) atomicAdd(&dst[lane], pts[(size_t)row * 3 + lane]);
  size_t rb = (size_t)row * Dd;
  #pragma unroll
  for (int j = 0; j < 8; ++j) {
    int d = lane * 8 + j;
    atomicAdd(&dst[3 + d], emb[rb + d]);
  }
}

// ---------- agg finalize: divide by count, zero-pad K 259 -> 288 -----------
__global__ void k_aggfin(const float* __restrict__ agg, const float* __restrict__ cnt,
                         float* __restrict__ aggp) {
  int i = blockIdx.x * 256 + threadIdx.x;
  if (i >= Ss * 288) return;
  int s = i / 288, c = i - s * 288;
  float v = 0.0f;
  if (c < 259) v = agg[(size_t)s * 260 + c] / fmaxf(cnt[s], 1.0f);
  aggp[i] = v;
}

// ---------- Wf1 zero-pad (259x256 -> 288x256) ------------------------------
__global__ void k_wf1pad(const float* __restrict__ Wf1, float* __restrict__ Wp) {
  int i = blockIdx.x * 256 + threadIdx.x;   // 288*256
  int r = i >> 8;
  Wp[i] = (r < 259) ? Wf1[i] : 0.0f;
}

// ---------- final: BN+GELU(y4) scattered into padded_feat, zeros elsewhere -
__global__ void k_featfin(const float* __restrict__ y4,
                          const float* __restrict__ g, const float* __restrict__ bi,
                          const float* __restrict__ sum, const float* __restrict__ sq,
                          float* __restrict__ out) {
  size_t i = (size_t)blockIdx.x * 256 + threadIdx.x;     // float4 index
  if (i >= (size_t)Bb * Nn * Dd / 4) return;
  size_t e = i * 4;
  int d0 = (int)(e & (Dd - 1));
  int n  = (int)((e >> 8) & (Nn - 1));
  int b  = (int)(e >> 22);
  float4 o = make_float4(0.f, 0.f, 0.f, 0.f);
  if (n < Vv) {
    int row = b * Vv + n;
    float4 y = *(const float4*)&y4[(size_t)row * Dd + d0];
    float vals[4] = { y.x, y.y, y.z, y.w };
    const float invM = 1.0f / (float)Ss;
    #pragma unroll
    for (int j = 0; j < 4; ++j) {
      int c = d0 + j;
      float m = sum[c] * invM;
      float v = sq[c] * invM - m * m;
      vals[j] = gelu_exact((vals[j] - m) * rsqrtf(v + 1e-5f) * g[c] + bi[c]);
    }
    o = make_float4(vals[0], vals[1], vals[2], vals[3]);
  }
  ((float4*)out)[i] = o;
}

// ---------- batch ids ------------------------------------------------------
__global__ void k_bid(float* __restrict__ out) {
  int i = blockIdx.x * 256 + threadIdx.x;
  if (i < BN_) out[i] = (float)(i >> 14);
}

// ===========================================================================
extern "C" void kernel_launch(void* const* d_in, const int* in_sizes, int n_in,
                              void* d_out, int out_size, void* d_ws, size_t ws_size,
                              hipStream_t stream) {
  const float* pts = (const float*)d_in[0];
  const float* W1  = (const float*)d_in[1];
  const float* g1  = (const float*)d_in[2];
  const float* b1  = (const float*)d_in[3];
  const float* W2  = (const float*)d_in[4];
  const float* g2  = (const float*)d_in[5];
  const float* b2  = (const float*)d_in[6];
  const float* Wf1 = (const float*)d_in[7];
  const float* gf1 = (const float*)d_in[8];
  const float* bf1 = (const float*)d_in[9];
  const float* Wf2 = (const float*)d_in[10];
  const float* gf2 = (const float*)d_in[11];
  const float* bf2 = (const float*)d_in[12];
  const int*   vox = (const int*)d_in[13];

  float* out  = (float*)d_out;
  float* ws   = (float*)d_ws;

  float* o_feat = out + OUT_FEAT;
  float* o_norm = out + OUT_NORM;
  float* o_cent = out + OUT_CENT;
  float* o_cnt  = out + OUT_CNT;
  float* o_emb  = out + OUT_EMB;   // holds y1, then y2, then final pos_embs
  float* o_bid  = out + OUT_BID;

  float* w_h1   = ws + WS_H1;
  float* w_aggp = ws + WS_AGGP;
  float* w_wf1p = ws + WS_WF1P;
  float* w_y3   = ws + WS_Y3;
  float* w_h2   = ws + WS_H2;
  float* w_agg  = ws + WS_AGG;
  float* w_y4   = ws + WS_Y4;
  float* st1s = ws + WS_ST;         float* st1q = st1s + 256;
  float* st2s = ws + WS_ST + 512;   float* st2q = st2s + 256;
  float* st3s = ws + WS_ST + 1024;  float* st3q = st3s + 256;
  float* st4s = ws + WS_ST + 1536;  float* st4q = st4s + 256;

  // zero accumulators (re-done every call; capture-safe)
  hipMemsetAsync(o_cent, 0, (size_t)Ss * 3 * sizeof(float), stream);
  hipMemsetAsync(o_cnt,  0, (size_t)Ss * sizeof(float), stream);
  hipMemsetAsync(w_agg,  0, (size_t)Ss * 260 * sizeof(float), stream);
  hipMemsetAsync(ws + WS_ST, 0, 2048 * sizeof(float), stream);

  // 1) segment count + xyz sums
  k_scatter1<<<BN_ / 256, 256, 0, stream>>>(pts, vox, o_cnt, o_cent);
  // 2) centroids
  k_centfin<<<Ss / 256, 256, 0, stream>>>(o_cent, o_cnt);
  // 3) norm points + padded_norm + y1 = norm @ W1  (y1 lives in pos_embs slot)
  k_norm_l1<<<BN_ / 8, 256, 0, stream>>>(pts, vox, o_cent, W1, o_norm, o_emb);
  // 4) BN1 stats + BN+GELU -> h1
  k_colstats<<<512, 256, 0, stream>>>(o_emb, BN_, st1s, st1q);
  k_bngelu<<<BN_ * 64 / 256, 256, 0, stream>>>(o_emb, w_h1, g1, b1, st1s, st1q, BN_);
  // 5) y2 = h1 @ W2 (WMMA, split-bf16)  -> overwrite pos_embs slot
  k_gemm<<<BN_ / 64, 128, 0, stream>>>(w_h1, W2, o_emb, Dd);
  // 6) BN2 stats + in-place BN+GELU -> final pos_embs
  k_colstats<<<512, 256, 0, stream>>>(o_emb, BN_, st2s, st2q);
  k_bngelu<<<BN_ * 64 / 256, 256, 0, stream>>>(o_emb, o_emb, g2, b2, st2s, st2q, BN_);
  // 7) segment sums of [xyz | pos_embs]
  k_scatter2<<<BN_ / 8, 256, 0, stream>>>(pts, vox, o_emb, w_agg);
  // 8) agg/cnt with K padded to 288; pad Wf1
  k_aggfin<<<(Ss * 288) / 256, 256, 0, stream>>>(w_agg, o_cnt, w_aggp);
  k_wf1pad<<<288, 256, 0, stream>>>(Wf1, w_wf1p);
  // 9) y3 = aggp @ Wf1p (K=288)
  k_gemm<<<Ss / 64, 128, 0, stream>>>(w_aggp, w_wf1p, w_y3, 288);
  // 10) BN f1 + GELU -> h2
  k_colstats<<<128, 256, 0, stream>>>(w_y3, Ss, st3s, st3q);
  k_bngelu<<<Ss * 64 / 256, 256, 0, stream>>>(w_y3, w_h2, gf1, bf1, st3s, st3q, Ss);
  // 11) y4 = h2 @ Wf2
  k_gemm<<<Ss / 64, 128, 0, stream>>>(w_h2, Wf2, w_y4, Dd);
  // 12) BN f2 stats; fused BN+GELU scatter into padded_feat (zeros for n>=V)
  k_colstats<<<128, 256, 0, stream>>>(w_y4, Ss, st4s, st4q);
  k_featfin<<<(Bb * Nn * Dd / 4) / 256, 256, 0, stream>>>(w_y4, gf2, bf2, st4s, st4q, o_feat);
  // 13) batch ids
  k_bid<<<BN_ / 256, 256, 0, stream>>>(o_bid);

  (void)in_sizes; (void)n_in; (void)out_size; (void)ws_size;
}
